// MultiHeadAttention_59579786330350
// MI455X (gfx1250) — matrix operand order repaired and checked by reading, not compiled, not run on hardware
//
#include <hip/hip_runtime.h>
#include <math.h>

// ---------------------------------------------------------------------------
// MultiHeadAttention for MI455X (gfx1250): f16 WMMA (v_wmma_f32_16x16x32_f16)
// B=4, T=1024, D=1024, H=16, DK=64
// ---------------------------------------------------------------------------

typedef __attribute__((ext_vector_type(16))) _Float16 v16h;
typedef __attribute__((ext_vector_type(8)))  _Float16 v8h;
typedef __attribute__((ext_vector_type(8)))  float    v8f;

constexpr int BB = 4;
constexpr int TT = 1024;
constexpr int DD = 1024;
constexpr int HH = 16;
constexpr int KK = 64;   // head dim

#define WMMA16(a, b, c) \
  __builtin_amdgcn_wmma_f32_16x16x32_f16(false, (a), false, (b), (short)0, (c), false, false)

// A-fragment loader for 16-bit 16x32 A tile:
// lane (half=lane>>4, m=lane&15): a[e] = A[m][K], K = e + 16*half.
// With p = &Arow[k0 + 16*half]: a[e] = p[e].
static __device__ __forceinline__ v16h load_a16(const _Float16* p) {
  v8h lo = *(const v8h*)(p);
  v8h hi = *(const v8h*)(p + 8);
  return __builtin_shufflevector(lo, hi, 0, 1, 2, 3, 4, 5, 6, 7,
                                         8, 9, 10, 11, 12, 13, 14, 15);
}

// ---------------------------------------------------------------------------
// Elementwise f32 -> f16
// ---------------------------------------------------------------------------
__global__ void f32_to_f16(const float* __restrict__ in, _Float16* __restrict__ out, int n) {
  int i = blockIdx.x * blockDim.x + threadIdx.x;
  if (i < n) out[i] = (_Float16)in[i];
}

// ---------------------------------------------------------------------------
// C[M,N] = A[M,Kd] @ W[N,Kd]^T + bias   (A,W f16 row-major; C f32)
// One wave computes a 32x64 tile: 2 A-frags x 4 B-frags per k-step
// -> 8 WMMA per 12 b128 loads (1.5 loads/WMMA vs 2.5 for 16x64 tiling).
// B-fragment (16-bit 32x16, contiguous k order as in load_a16): lane n=lane&15 holds column n,
// b[e] = B[e + 16*half][n] = W[n][k0 + 16*half + e]  -> 16 contiguous f16.
// ---------------------------------------------------------------------------
__global__ __launch_bounds__(256)
void gemm16(const _Float16* __restrict__ A, const _Float16* __restrict__ W,
            const float* __restrict__ bias, float* __restrict__ C,
            int M, int N, int Kd) {
  const int lane = threadIdx.x & 31;
  const int wave = threadIdx.x >> 5;
  const int tilesN = N >> 6;
  const int tilesM = M >> 5;                 // 32-row tiles
  const int tile = blockIdx.x * 8 + wave;
  if (tile >= tilesM * tilesN) return;
  const int tm = tile / tilesN;
  const int tn = tile % tilesN;
  const int hlf = lane >> 4;
  const int ln  = lane & 15;

  v8f acc0[4] = {};
  v8f acc1[4] = {};
  const _Float16* arow0 = A + (size_t)(tm * 32 + ln) * Kd + 16 * hlf;
  const _Float16* arow1 = arow0 + (size_t)16 * Kd;
  const _Float16* wrow  = W + (size_t)(tn * 64 + ln) * Kd + 16 * hlf;

  for (int k0 = 0; k0 < Kd; k0 += 32) {
    v16h a0 = load_a16(arow0 + k0);
    v16h a1 = load_a16(arow1 + k0);
#pragma unroll
    for (int j = 0; j < 4; ++j) {
      v16h b = *(const v16h*)(wrow + k0 + (size_t)(16 * j) * Kd);
      acc0[j] = WMMA16(a0, b, acc0[j]);
      acc1[j] = WMMA16(a1, b, acc1[j]);
    }
  }

  // C/D layout (ISA 7.12.2): lane holds col n=ln, rows r + 8*half, r=0..7.
#pragma unroll
  for (int j = 0; j < 4; ++j) {
    const int col = tn * 64 + j * 16 + ln;
    const float bv = bias[col];
#pragma unroll
    for (int r = 0; r < 8; ++r) {
      C[(size_t)(tm * 32 + r + 8 * hlf) * N + col]      = acc0[j][r] + bv;
      C[(size_t)(tm * 32 + 16 + r + 8 * hlf) * N + col] = acc1[j][r] + bv;
    }
  }
}

// ---------------------------------------------------------------------------
// Per-head LayerNorm over DK=64: proj f32 [B,T,H*DK] -> out f16 [B,H,T,DK]
// One 64-thread block per (b,t,h).
// ---------------------------------------------------------------------------
__global__ __launch_bounds__(64)
void ln_qk(const float* __restrict__ proj, const float* __restrict__ gamma,
           const float* __restrict__ beta, _Float16* __restrict__ out) {
  const int gid = blockIdx.x;
  const int h = gid % HH;
  const int t = (gid / HH) % TT;
  const int b = gid / (HH * TT);
  const int d = threadIdx.x;

  __shared__ float s1[64], s2[64];
  float x = proj[((size_t)(b * TT + t)) * DD + h * KK + d];
  s1[d] = x;
  s2[d] = x * x;
  __syncthreads();
  for (int off = 32; off > 0; off >>= 1) {
    if (d < off) { s1[d] += s1[d + off]; s2[d] += s2[d + off]; }
    __syncthreads();
  }
  float mu  = s1[0] * (1.0f / 64.0f);
  float var = s2[0] * (1.0f / 64.0f) - mu * mu;
  float y = (x - mu) * rsqrtf(var + 1e-5f) * gamma[d] + beta[d];
  out[(((size_t)(b * HH + h)) * TT + t) * KK + d] = (_Float16)y;
}

// ---------------------------------------------------------------------------
// V reshape: proj f32 [B,T,H*DK] -> Vt f16 [B,H,DK,T] (d-major so the P@V
// B-fragment reads 16 contiguous t-values per lane).
// ---------------------------------------------------------------------------
__global__ void v_heads(const float* __restrict__ proj, _Float16* __restrict__ Vt) {
  int i = blockIdx.x * blockDim.x + threadIdx.x;
  if (i >= BB * TT * DD) return;
  int d = i % KK;
  int h = (i / KK) % HH;
  int t = (i / DD) % TT;
  int b = i / (TT * DD);
  Vt[(((size_t)(b * HH + h)) * KK + d) * TT + t] = (_Float16)proj[i];
}

// ---------------------------------------------------------------------------
// Flash-style attention. One wave per (b,h, 16-row q tile).
// Per 32-col k block: 4 WMMA for S = Q K^T, LDS-staged online softmax
// (mask -> -10000, masked p -> 0, matching reference), 4 WMMA for P V.
// ---------------------------------------------------------------------------
__global__ __launch_bounds__(32)
void attn16(const _Float16* __restrict__ Q, const _Float16* __restrict__ K,
            const _Float16* __restrict__ V, const unsigned char* __restrict__ mask,
            _Float16* __restrict__ ctx) {
  const int lane = threadIdx.x;
  const int hlf = lane >> 4;
  const int ln  = lane & 15;
  const int qt = blockIdx.x;        // q tile (16 rows)
  const int bh = blockIdx.y;        // b*H + h
  const int b  = bh >> 4;           // HH == 16
  const int h  = bh & 15;

  __shared__ float    sS[16 * 32];
  __shared__ _Float16 sP[16 * 32];
  __shared__ float    sM[16], sL[16], sAl[16];

  if (lane < 16) { sM[lane] = -3.0e38f; sL[lane] = 0.0f; }
  __syncthreads();

  // Q A-fragments (16 x 64 -> two 16x32 frags), loaded once.
  const _Float16* qrow = Q + ((size_t)bh * TT + qt * 16 + ln) * KK + 16 * hlf;
  v16h aq0 = load_a16(qrow);
  v16h aq1 = load_a16(qrow + 32);

  v8f o[4] = {};

  for (int kb = 0; kb < TT / 32; ++kb) {
    // ---- scores: S(16x32) = Q(16x64) @ K_block(32x64)^T ----
#pragma unroll
    for (int st = 0; st < 2; ++st) {
      const _Float16* krow =
          K + ((size_t)bh * TT + kb * 32 + st * 16 + ln) * KK + 16 * hlf;
      v16h b0 = *(const v16h*)(krow);        // d in [0,32)
      v16h b1 = *(const v16h*)(krow + 32);   // d in [32,64)
      v8f s = {};
      s = WMMA16(aq0, b0, s);
      s = WMMA16(aq1, b1, s);
#pragma unroll
      for (int r = 0; r < 8; ++r)
        sS[(r + 8 * hlf) * 32 + st * 16 + ln] = s[r];
    }
    __syncthreads();

    // ---- online softmax (lanes 0..15, one q row each) ----
    if (lane < 16) {
      const unsigned char* mrow =
          mask + ((size_t)b * TT + qt * 16 + lane) * TT + kb * 32;
      float mOld = sM[lane];
      float mNew = mOld;
      float sv[32];
#pragma unroll
      for (int c = 0; c < 32; ++c) {
        float s = sS[lane * 32 + c] * 0.125f;       // 1/sqrt(64)
        s = mrow[c] ? -10000.0f : s;                // reference masking
        sv[c] = s;
        mNew = fmaxf(mNew, s);
      }
      float sum = 0.0f;
#pragma unroll
      for (int c = 0; c < 32; ++c) {
        float p = mrow[c] ? 0.0f : __expf(sv[c] - mNew);  // attn zeroed on mask
        sum += p;
        sP[lane * 32 + c] = (_Float16)p;
      }
      float al = __expf(mOld - mNew);
      sAl[lane] = al;
      sL[lane] = sL[lane] * al + sum;
      sM[lane] = mNew;
    }
    __syncthreads();

    // ---- rescale accumulators, O += P @ V_block ----
    float al[8];
#pragma unroll
    for (int r = 0; r < 8; ++r) al[r] = sAl[r + 8 * hlf];
#pragma unroll
    for (int j = 0; j < 4; ++j) {
#pragma unroll
      for (int r = 0; r < 8; ++r) o[j][r] *= al[r];
    }

    v16h ap = load_a16(&sP[ln * 32 + 16 * hlf]);  // P as A-fragment from LDS
    const _Float16* vcol =
        V + ((size_t)bh * KK + ln) * TT + kb * 32 + 16 * hlf;
#pragma unroll
    for (int j = 0; j < 4; ++j) {
      v16h bv = *(const v16h*)(vcol + (size_t)(16 * j) * TT);
      o[j] = WMMA16(ap, bv, o[j]);
    }
  }

  // ---- epilogue: divide by l (guard fully-masked rows), write ctx ----
  float l[8];
#pragma unroll
  for (int r = 0; r < 8; ++r) l[r] = sL[r + 8 * hlf];
#pragma unroll
  for (int j = 0; j < 4; ++j) {
#pragma unroll
    for (int r = 0; r < 8; ++r) {
      float valr = (l[r] > 0.0f) ? o[j][r] / l[r] : 0.0f;
      ctx[((size_t)(b * TT + qt * 16 + r + 8 * hlf)) * DD + h * KK + j * 16 + ln] =
          (_Float16)valr;
    }
  }
}

// ---------------------------------------------------------------------------
// Host-side launcher
// ---------------------------------------------------------------------------
extern "C" void kernel_launch(void* const* d_in, const int* in_sizes, int n_in,
                              void* d_out, int out_size, void* d_ws, size_t ws_size,
                              hipStream_t stream) {
  (void)in_sizes; (void)n_in; (void)out_size; (void)ws_size;

  const float* query = (const float*)d_in[0];
  const float* key   = (const float*)d_in[1];
  const float* value = (const float*)d_in[2];
  const unsigned char* mask = (const unsigned char*)d_in[3];  // numpy bool = 1 byte
  const float* Wq = (const float*)d_in[4];
  const float* bq = (const float*)d_in[5];
  const float* Wk = (const float*)d_in[6];
  const float* bk = (const float*)d_in[7];
  const float* Wv = (const float*)d_in[8];
  const float* bv = (const float*)d_in[9];
  const float* Wo = (const float*)d_in[10];
  const float* bo = (const float*)d_in[11];
  const float* qg = (const float*)d_in[12];
  const float* qb = (const float*)d_in[13];
  const float* kg = (const float*)d_in[14];
  const float* kbeta = (const float*)d_in[15];
  float* out = (float*)d_out;

  const int    MT = BB * TT;              // 4096 rows
  const size_t ND = (size_t)MT * DD;      // 4M activation elems
  const size_t NW = (size_t)DD * DD;      // 1M weight elems

  char* ws = (char*)d_ws;
  size_t off = 0;
  auto alloc = [&](size_t bytes) -> char* {
    char* p = ws + off;
    off += (bytes + 255) & ~(size_t)255;
    return p;
  };

  _Float16* Xq  = (_Float16*)alloc(ND * 2);
  _Float16* Xk  = (_Float16*)alloc(ND * 2);
  _Float16* Xv  = (_Float16*)alloc(ND * 2);
  _Float16* Wqh = (_Float16*)alloc(NW * 2);
  _Float16* Wkh = (_Float16*)alloc(NW * 2);
  _Float16* Wvh = (_Float16*)alloc(NW * 2);
  _Float16* Woh = (_Float16*)alloc(NW * 2);
  float*    Pq  = (float*)alloc(ND * 4);
  float*    Pk  = (float*)alloc(ND * 4);
  float*    Pv  = (float*)alloc(ND * 4);
  _Float16* Qh  = (_Float16*)alloc(ND * 2);   // [B,H,T,DK] layernormed
  _Float16* Kh  = (_Float16*)alloc(ND * 2);   // [B,H,T,DK] layernormed
  _Float16* Vt  = (_Float16*)alloc(ND * 2);   // [B,H,DK,T]
  _Float16* ctx = (_Float16*)Pq;              // alias: Pq is dead after ln_qk

  const dim3 b256(256);
  const int gridND = (int)((ND + 255) / 256);
  const int gridNW = (int)((NW + 255) / 256);

  // 1) f32 -> f16 conversions
  f32_to_f16<<<gridND, b256, 0, stream>>>(query, Xq, (int)ND);
  f32_to_f16<<<gridND, b256, 0, stream>>>(key,   Xk, (int)ND);
  f32_to_f16<<<gridND, b256, 0, stream>>>(value, Xv, (int)ND);
  f32_to_f16<<<gridNW, b256, 0, stream>>>(Wq, Wqh, (int)NW);
  f32_to_f16<<<gridNW, b256, 0, stream>>>(Wk, Wkh, (int)NW);
  f32_to_f16<<<gridNW, b256, 0, stream>>>(Wv, Wvh, (int)NW);
  f32_to_f16<<<gridNW, b256, 0, stream>>>(Wo, Woh, (int)NW);

  // 2) Q/K/V projections (WMMA GEMM): 2048 tiles of 32x64, 8 waves/block
  const int tiles   = (MT / 32) * (DD / 64);
  const int gblocks = (tiles + 7) / 8;
  gemm16<<<gblocks, b256, 0, stream>>>(Xq, Wqh, bq, Pq, MT, DD, DD);
  gemm16<<<gblocks, b256, 0, stream>>>(Xk, Wkh, bk, Pk, MT, DD, DD);
  gemm16<<<gblocks, b256, 0, stream>>>(Xv, Wvh, bv, Pv, MT, DD, DD);

  // 3) per-head layernorm + head reshape
  ln_qk<<<BB * TT * HH, 64, 0, stream>>>(Pq, qg, qb, Qh);
  ln_qk<<<BB * TT * HH, 64, 0, stream>>>(Pk, kg, kbeta, Kh);
  v_heads<<<gridND, b256, 0, stream>>>(Pv, Vt);

  // 4) flash attention (WMMA QK^T + online softmax + WMMA PV)
  dim3 agrid(TT / 16, BB * HH);
  attn16<<<agrid, 32, 0, stream>>>(Qh, Kh, Vt, mask, ctx);

  // 5) output projection -> f32 d_out
  gemm16<<<gblocks, b256, 0, stream>>>(ctx, Woh, bo, out, MT, DD, DD);
}
